// ProjectiveLinear_37177236914407
// MI455X (gfx1250) — matrix operand order repaired and checked by reading, not compiled, not run on hardware
//
#include <hip/hip_runtime.h>

typedef __attribute__((ext_vector_type(16))) _Float16 v16h;
typedef __attribute__((ext_vector_type(8)))  _Float16 v8h;
typedef __attribute__((ext_vector_type(4)))  _Float16 v4h;
typedef __attribute__((ext_vector_type(8)))  float    v8f;
typedef __attribute__((ext_vector_type(4)))  float    v4f;
typedef __attribute__((ext_vector_type(4)))  int      v4i;

// Problem dims (from reference)
constexpr int M = 4096;    // BATCH * SEQ
constexpr int N = 16384;   // OUT_FEATURES
constexpr int K = 4096;    // IN_FEATURES

// GEMM tiling
constexpr int BM   = 128;  // block tile M
constexpr int BN   = 128;  // block tile N
constexpr int KB   = 32;   // K per LDS stage (== WMMA K)
constexpr int KBP  = 40;   // padded row stride in halves (80B)
constexpr int NBUF = 3;    // async pipeline depth (3 x 20KB = 60KB LDS)

// ---------------------------------------------------------------------------
// Pre-pass 1: convert x (fp32) -> fp16
// ---------------------------------------------------------------------------
__global__ __launch_bounds__(256) void cvt_x_kernel(const float* __restrict__ x,
                                                    _Float16* __restrict__ xh) {
  size_t i = ((size_t)blockIdx.x * 256 + threadIdx.x) * 8;
  v4f a = *(const v4f*)(x + i);
  v4f b = *(const v4f*)(x + i + 4);
  v8h h;
#pragma unroll
  for (int j = 0; j < 4; ++j) {
    h[j]     = (_Float16)a[j];
    h[j + 4] = (_Float16)b[j];
  }
  *(v8h*)(xh + i) = h;
}

// ---------------------------------------------------------------------------
// Pre-pass 2: dequantize log-quantized weights -> fp16
//   w = (sign ? +1 : -1) * exp(lmin + (255 - stored)/254 * (lmax - lmin))
// ---------------------------------------------------------------------------
__global__ __launch_bounds__(256) void dequant_kernel(const int* __restrict__ stored,
                                                      const int* __restrict__ sign,
                                                      const float* __restrict__ lminp,
                                                      const float* __restrict__ lmaxp,
                                                      _Float16* __restrict__ wh) {
  size_t i = ((size_t)blockIdx.x * 256 + threadIdx.x) * 4;
  v4i s = *(const v4i*)(stored + i);
  v4i g = *(const v4i*)(sign + i);
  float lmin  = lminp[0];
  float scale = (lmaxp[0] - lmin) * (1.0f / 254.0f);
  v4h h;
#pragma unroll
  for (int j = 0; j < 4; ++j) {
    float lw  = lmin + (float)(255 - s[j]) * scale;
    float mag = __expf(lw);
    h[j] = (_Float16)(g[j] ? mag : -mag);
  }
  *(v4h*)(wh + i) = h;
}

// ---------------------------------------------------------------------------
// CDNA5 async copy: global -> LDS, 16B per lane, ASYNCcnt-tracked.
// GVS mode: addr = SGPR64(base) + VGPR32(byte offset). VDST = LDS byte offset.
// ---------------------------------------------------------------------------
__device__ __forceinline__ void async_copy_b128(unsigned lds_off, unsigned gvoff,
                                                const _Float16* base) {
  asm volatile("global_load_async_to_lds_b128 %0, %1, %2"
               :: "v"(lds_off), "v"(gvoff), "s"(base)
               : "memory");
}

// Low 32 bits of a generic pointer to __shared__ data == LDS byte offset.
__device__ __forceinline__ unsigned lds_addr_of(const void* p) {
  return (unsigned)(uintptr_t)p;
}

// ---------------------------------------------------------------------------
// WMMA GEMM: out[M,N] = xh[M,K] * wh[N,K]^T + bias
// 256 threads = 8 waves (wave32); block tile 128x128; waves 4(M) x 2(N),
// each wave computes 32x64 = 2x4 WMMA 16x16x32 tiles.
// 3-deep async global->LDS pipeline, one barrier per K-stage.
// ---------------------------------------------------------------------------
__global__ __launch_bounds__(256) void wmma_gemm_kernel(const _Float16* __restrict__ xh,
                                                        const _Float16* __restrict__ wh,
                                                        const float* __restrict__ bias,
                                                        float* __restrict__ out) {
  __shared__ __align__(16) _Float16 sA[NBUF][BM][KBP];
  __shared__ __align__(16) _Float16 sB[NBUF][BN][KBP];

  const int t    = threadIdx.x;
  const int lane = t & 31;
  const int wave = t >> 5;

  const int blockMBase = blockIdx.y * BM;
  const int blockNBase = blockIdx.x * BN;

  const int mW = (wave & 3) * 32;  // wave's M offset in block tile
  const int nW = (wave >> 2) * 64; // wave's N offset in block tile

  // Staging map: 256 threads cover 128 rows x 32 halves (64B) per tile;
  // each thread owns 32B = two async b128 copies per tile per stage.
  const int lrow = t >> 1;
  const int lcol = (t & 1) * 16;  // halves

  // Global byte offsets (GVS VGPR operand), advance by KB*2 bytes per stage.
  const unsigned gvA = (unsigned)(((blockMBase + lrow) * K + lcol) * 2);
  const unsigned gvB = (unsigned)(((blockNBase + lrow) * K + lcol) * 2);

  constexpr int NS = K / KB;  // 128 stages

  auto issue_stage = [&](int ks) {
    const int b = ks % NBUF;
    const unsigned la = lds_addr_of(&sA[b][lrow][lcol]);
    const unsigned lb = lds_addr_of(&sB[b][lrow][lcol]);
    const unsigned go = (unsigned)(ks * KB * 2);
    async_copy_b128(la,      gvA + go,      xh);
    async_copy_b128(la + 16, gvA + go + 16, xh);
    async_copy_b128(lb,      gvB + go,      wh);
    async_copy_b128(lb + 16, gvB + go + 16, wh);
  };

  v8f acc[2][4];
#pragma unroll
  for (int i = 0; i < 2; ++i)
#pragma unroll
    for (int j = 0; j < 4; ++j) acc[i][j] = {};

  // Prologue: fill pipeline with stages 0 and 1.
  issue_stage(0);
  issue_stage(1);
  asm volatile("s_wait_asynccnt 0x4" ::: "memory");  // stage 0 landed
  __syncthreads();

  // Fragment addressing (CDNA5 16-bit WMMA VGPR layouts):
  // A 16x32: lane<16 -> row=lane,    K chunks {0..7, 16..23}
  //          lane>=16 -> row=lane-16, K chunks {8..15, 24..31}
  // B 32x16: lane<16 -> col=lane, K {0..7, 16..23} ; lane>=16 -> col=lane-16, K {8..15, 24..31}
  const int arow = lane & 15;
  const int hi   = lane >> 4;

  union Frag {
    v16h v;
    v4f  f[2];
  };

  for (int ks = 0; ks < NS; ++ks) {
    const int buf = ks % NBUF;

    // Keep the pipeline full: stage ks+2 lands while we compute stage ks.
    // Writing buf (ks+2)%NBUF == (ks-1)%NBUF is safe: its readers finished
    // before the barrier that ended iteration ks-1.
    if (ks + 2 < NS) issue_stage(ks + 2);

    Frag afrag[2], bfrag[4];
#pragma unroll
    for (int tm = 0; tm < 2; ++tm) {
      const _Float16* p = &sA[buf][mW + tm * 16 + arow][hi * 8];
      afrag[tm].f[0] = *(const v4f*)(p);        // K chunk 0 / 8
      afrag[tm].f[1] = *(const v4f*)(p + 16);   // K chunk 16 / 24
    }
#pragma unroll
    for (int tn = 0; tn < 4; ++tn) {
      const _Float16* p = &sB[buf][nW + tn * 16 + arow][hi * 8];
      bfrag[tn].f[0] = *(const v4f*)(p);        // K 0..7  / 8..15
      bfrag[tn].f[1] = *(const v4f*)(p + 16);   // K 16..23 / 24..31
    }

#pragma unroll
    for (int tm = 0; tm < 2; ++tm)
#pragma unroll
      for (int tn = 0; tn < 4; ++tn)
        acc[tm][tn] = __builtin_amdgcn_wmma_f32_16x16x32_f16(
            false, afrag[tm].v, false, bfrag[tn].v, (short)0, acc[tm][tn],
            false, false);

    // Ensure stage ks+1 has landed in LDS (in-order completion: leaving the
    // 4 just-issued stage-(ks+2) copies outstanding proves ks+1 is done),
    // then publish across waves.
    if (ks + 2 < NS) {
      asm volatile("s_wait_asynccnt 0x4" ::: "memory");
    } else if (ks + 1 < NS) {
      asm volatile("s_wait_asynccnt 0x0" ::: "memory");
    }
    __syncthreads();
  }

  // Epilogue: C 16x16 f32 layout -> VGPR r holds rows {r, r+8}, lane&15 = col.
#pragma unroll
  for (int tn = 0; tn < 4; ++tn) {
    const int   col = blockNBase + nW + tn * 16 + arow;
    const float bv  = bias[col];
#pragma unroll
    for (int tm = 0; tm < 2; ++tm) {
      const int row0 = blockMBase + mW + tm * 16 + hi * 8;
#pragma unroll
      for (int r = 0; r < 8; ++r) {
        out[(size_t)(row0 + r) * N + col] = acc[tm][tn][r] + bv;
      }
    }
  }
}

// ---------------------------------------------------------------------------
// Host launcher
// ---------------------------------------------------------------------------
extern "C" void kernel_launch(void* const* d_in, const int* in_sizes, int n_in,
                              void* d_out, int out_size, void* d_ws, size_t ws_size,
                              hipStream_t stream) {
  (void)in_sizes; (void)n_in; (void)out_size; (void)ws_size;

  const float* x      = (const float*)d_in[0];
  const int*   stored = (const int*)d_in[1];
  const int*   sign   = (const int*)d_in[2];
  const float* lmin   = (const float*)d_in[3];
  const float* lmax   = (const float*)d_in[4];
  const float* bias   = (const float*)d_in[5];
  float*       out    = (float*)d_out;

  // Workspace layout: [ xh: M*K fp16 = 32MB | wh: N*K fp16 = 128MB ]
  _Float16* xh = (_Float16*)d_ws;
  _Float16* wh = (_Float16*)((char*)d_ws + (size_t)M * K * sizeof(_Float16));

  // x fp32 -> fp16 (8 elements/thread)
  cvt_x_kernel<<<((size_t)M * K) / (256 * 8), 256, 0, stream>>>(x, xh);

  // weight dequant (4 elements/thread)
  dequant_kernel<<<((size_t)N * K) / (256 * 4), 256, 0, stream>>>(stored, sign,
                                                                  lmin, lmax, wh);

  // WMMA GEMM
  dim3 grid(N / BN, M / BM);
  wmma_gemm_kernel<<<grid, 256, 0, stream>>>(xh, wh, bias, out);
}
